// Low_Decoder_27419071218003
// MI455X (gfx1250) — matrix operand order repaired and checked by reading, not compiled, not run on hardware
//
#include <hip/hip_runtime.h>
#include <hip/hip_bf16.h>
#include <math.h>

// Problem dims from the reference: B=512, S=256, E=H=128 (K=N=128 fixed GEMMs)
#define BDIM 512
#define SDIM 256
#define EDIM 128
#define HDIM 128
#define NEG_INFF (-1e8f)
#define C_LOGIT 10.0f

typedef __attribute__((ext_vector_type(16))) __bf16 v16bf;
typedef __attribute__((ext_vector_type(8)))  __bf16 v8bf;
typedef __attribute__((ext_vector_type(4)))  __bf16 v4bf;
typedef __attribute__((ext_vector_type(8)))  float  v8f;
typedef __attribute__((ext_vector_type(4)))  unsigned u32x4;
typedef __attribute__((ext_vector_type(8)))  int      i32x8;
typedef __attribute__((ext_vector_type(4)))  int      i32x4;

// f32 -> bf16 round-to-nearest-even (bit-level)
__device__ __forceinline__ __bf16 f2bf(float x) {
    unsigned u = __float_as_uint(x);
    unsigned r = (u + 0x7FFFu + ((u >> 16) & 1u)) >> 16;
    return __builtin_bit_cast(__bf16, (unsigned short)r);
}

// hardware v_tanh_f32 if the builtin exists, else libm
__device__ __forceinline__ float fast_tanh(float x) {
#if __has_builtin(__builtin_amdgcn_tanhf)
    return __builtin_amdgcn_tanhf(x);
#elif __has_builtin(__builtin_amdgcn_tanh_f32)
    return __builtin_amdgcn_tanh_f32(x);
#else
    return tanhf(x);
#endif
}

// ---------------------------------------------------------------------------
// h_bar[b,:] = mean_s(cv[b,s,:]) @ Wc + bc         (grid = B, block = 128)
// ---------------------------------------------------------------------------
__global__ void k_mean_hbar(const float* __restrict__ cv,
                            const float* __restrict__ Wc,
                            const float* __restrict__ bc,
                            float* __restrict__ hbar) {
    __shared__ float m[EDIM];
    int b = blockIdx.x, e = threadIdx.x;
    const float* p = cv + (size_t)b * SDIM * EDIM + e;
    float acc = 0.f;
    for (int s = 0; s < SDIM; ++s) acc += p[(size_t)s * EDIM];
    m[e] = acc * (1.0f / SDIM);
    __syncthreads();
    float o = bc[e];
    for (int k = 0; k < EDIM; ++k) o = fmaf(m[k], Wc[k * EDIM + e], o);
    hbar[(size_t)b * EDIM + e] = o;
}

// ---------------------------------------------------------------------------
// init: mask copy (mask[:,0]=1), query0 (bf16) = h_bar + (liw@Wv + bv),
// init_h = 0.                                      (grid = B, block = 256)
// ---------------------------------------------------------------------------
__global__ void k_init(const float* __restrict__ mask_in,
                       const float* __restrict__ liw,
                       const float* __restrict__ Wv,
                       const float* __restrict__ bv,
                       const float* __restrict__ hbar,
                       float* __restrict__ maskw,
                       __bf16* __restrict__ qbf,
                       float* __restrict__ inith) {
    __shared__ float r0[EDIM];
    int b = blockIdx.x, t = threadIdx.x;
    if (t < EDIM) {
        float acc = bv[t];
        for (int j = 0; j < 2 * EDIM; ++j) acc = fmaf(liw[j], Wv[j * EDIM + t], acc);
        r0[t] = acc;
    }
    maskw[(size_t)b * SDIM + t] = (t == 0) ? 1.0f : mask_in[(size_t)b * SDIM + t];
    __syncthreads();
    if (t < EDIM) {
        qbf[(size_t)b * EDIM + t]   = f2bf(hbar[(size_t)b * EDIM + t] + r0[t]);
        inith[(size_t)b * EDIM + t] = 0.f;
    }
}

// ---------------------------------------------------------------------------
// cv (f32) -> cvbf (bf16), vectorized b128 in / b64 out. n % 4 == 0.
// ---------------------------------------------------------------------------
__global__ void k_cvt_bf16(const float4* __restrict__ in, v4bf* __restrict__ out) {
    size_t i = (size_t)blockIdx.x * blockDim.x + threadIdx.x;
    float4 f = in[i];
    v4bf o = { f2bf(f.x), f2bf(f.y), f2bf(f.z), f2bf(f.w) };
    out[i] = o;
}

// ---------------------------------------------------------------------------
// Pack W[128,128] (f32, row-major KxN) into per-lane WMMA B-fragment order:
//   frag[((chunk*8+nt)*32 + lane)*16 + j] = bf16(W[(chunk*32+kb+j)*128 + nt*16+(lane&15)])
//   kb = (lane<16) ? 0 : 16   (contiguous k = 16h + j, same order as the A loads)
// 16384 elements total -> 64 blocks x 256.
// ---------------------------------------------------------------------------
__global__ void k_packW(const float* __restrict__ W, __bf16* __restrict__ frag) {
    int i = blockIdx.x * blockDim.x + threadIdx.x;   // 0..16383
    int j     = i & 15;
    int lane  = (i >> 4) & 31;
    int nt    = (i >> 9) & 7;
    int chunk = i >> 12;
    int kb = (lane < 16) ? 0 : 16;
    int k  = chunk * 32 + kb + j;
    int n  = nt * 16 + (lane & 15);
    frag[i] = f2bf(W[k * HDIM + n]);
}

// ---------------------------------------------------------------------------
// C[M,128] = A[M,128](bf16) @ W (pre-swizzled 32KB fragment block).
// Block = 256 threads = 8 waves; wave -> 16x128 strip; 8 f32 accumulators;
// 4 K-chunks x 8 N-tiles = 32 unrolled v_wmma per wave.
// W fragments staged to LDS via TDM (tensor_load_to_lds) when available.
// Requires M % 128 == 0.
// ---------------------------------------------------------------------------
__global__ void k_gemm_bf16(const __bf16* __restrict__ A,
                            const __bf16* __restrict__ Wfrag,
                            float* __restrict__ C, int M) {
    __shared__ __bf16 ldsB[4 * 8 * 32 * 16];   // 32 KB
    int lane = threadIdx.x & 31;
    int wave = threadIdx.x >> 5;

#if __has_builtin(__builtin_amdgcn_tensor_load_to_lds)
    if (wave == 0) {
        // Tensor DMA descriptor (ISA ch.8): 1-D tile, 16384 x 2B elements.
        unsigned long long ga = (unsigned long long)(size_t)Wfrag;
        unsigned lds_base = (unsigned)(size_t)&ldsB[0];
        u32x4 g0 = { 1u,                                  // count=1, user desc
                     lds_base,                            // lds_addr
                     (unsigned)(ga & 0xFFFFFFFFu),        // global_addr[31:0]
                     (unsigned)((ga >> 32) & 0x1FFFFFFu)  // global_addr[56:32]
                       | 0x80000000u };                   // type=2 ("image")
        i32x8 g1 = { (int)0x00010000,   // wg_mask=0, data_size=1 (2B)
                     (int)0x40000000,   // tensor_dim0[15:0]=16384 << 16
                     (int)0x00010000,   // tensor_dim0 hi=0 | tensor_dim1=1 << 16
                     (int)0x40000000,   // tensor_dim1 hi=0 | tile_dim0=16384 << 16
                     1,                 // tile_dim1=1 | tile_dim2=0
                     16384,             // tensor_dim0_stride[31:0]
                     0, 0 };
        i32x4 g2 = { 0, 0, 0, 0 }, g3 = { 0, 0, 0, 0 };
        i32x8 g4 = { 0, 0, 0, 0, 0, 0, 0, 0 };           // unused 5th group
        __builtin_amdgcn_tensor_load_to_lds(g0, g1, g2, g3, g4, 0);
    }
    __builtin_amdgcn_s_wait_tensorcnt(0);
    __syncthreads();
#else
    for (int it = threadIdx.x; it < 2048; it += 256)     // 2048 x 16B = 32 KB
        *(v8bf*)(ldsB + it * 8) = *(const v8bf*)(Wfrag + it * 8);
    __syncthreads();
#endif

    int row0 = (blockIdx.x * 8 + wave) * 16;
    int m  = row0 + (lane & 15);
    int ka = (lane < 16) ? 0 : 16;         // contiguous k = 16h + s, matches packed B
    const __bf16* pA = A + (size_t)m * EDIM;

    v8f c[8] = {};
#pragma unroll
    for (int chunk = 0; chunk < 4; ++chunk) {
        const __bf16* pa = pA + chunk * 32 + ka;
        v8bf a0 = *(const v8bf*)(pa);
        v8bf a1 = *(const v8bf*)(pa + 8);
        v16bf a = __builtin_shufflevector(a0, a1, 0, 1, 2, 3, 4, 5, 6, 7,
                                                  8, 9, 10, 11, 12, 13, 14, 15);
#pragma unroll
        for (int nt = 0; nt < 8; ++nt) {
            const __bf16* pb = ldsB + (((chunk << 3) | nt) << 9) + (lane << 4);
            v8bf b0 = *(const v8bf*)(pb);
            v8bf b1 = *(const v8bf*)(pb + 8);
            v16bf bb = __builtin_shufflevector(b0, b1, 0, 1, 2, 3, 4, 5, 6, 7,
                                                       8, 9, 10, 11, 12, 13, 14, 15);
            c[nt] = __builtin_amdgcn_wmma_f32_16x16x32_bf16(
                        false, a, false, bb, (short)0, c[nt], false, false);
        }
    }
    int mb = (lane < 16) ? 0 : 8;
    int nn = lane & 15;
#pragma unroll
    for (int nt = 0; nt < 8; ++nt)
#pragma unroll
        for (int i = 0; i < 8; ++i)
            C[(size_t)(row0 + mb + i) * HDIM + nt * 16 + nn] = c[nt][i];
}

// ---------------------------------------------------------------------------
// One decode step per (block = batch element b, 256 threads = S):
//   u[s] = sum_h tanh(qW[b,h] + ref[b,s,h]) * v[h]   (hardware tanh, b128 ref)
//   masked logits -> log-softmax -> Gumbel argmax; update mask/init_h/query.
// ---------------------------------------------------------------------------
__global__ void k_step(const float* __restrict__ cv,
                       const float* __restrict__ ref,
                       const float* __restrict__ qW,
                       const float* __restrict__ vvec,
                       const float* __restrict__ hbar,
                       const float* __restrict__ Wv,
                       const float* __restrict__ bv,
                       float* __restrict__ maskw,
                       __bf16* __restrict__ qbf,
                       float* __restrict__ inith,
                       float* __restrict__ out_idx,
                       float* __restrict__ out_lp,
                       int step) {
    __shared__ float qrow[HDIM], vv[HDIM], logits[SDIM];
    __shared__ float red[SDIM];
    __shared__ int   redi[SDIM];
    __shared__ float hsel[EDIM], ihrow[EDIM];
    __shared__ int   sel;

    int b = blockIdx.x, t = threadIdx.x;
    const float* r = ref + ((size_t)b * SDIM + t) * HDIM;
    __builtin_prefetch(r, 0, 3);                  // global_prefetch_b8
    if (t < HDIM) { qrow[t] = qW[(size_t)b * HDIM + t]; vv[t] = vvec[t]; }
    __syncthreads();

    // ---- scores: one ref row per thread (L2-resident: 64MB < 192MB L2)
    {
        const float4* r4 = (const float4*)r;
        float acc = 0.f;
#pragma unroll
        for (int h4 = 0; h4 < HDIM / 4; ++h4) {
            float4 rv = r4[h4];
            int h = h4 * 4;
            acc = fmaf(fast_tanh(qrow[h + 0] + rv.x), vv[h + 0], acc);
            acc = fmaf(fast_tanh(qrow[h + 1] + rv.y), vv[h + 1], acc);
            acc = fmaf(fast_tanh(qrow[h + 2] + rv.z), vv[h + 2], acc);
            acc = fmaf(fast_tanh(qrow[h + 3] + rv.w), vv[h + 3], acc);
        }
        float lg = C_LOGIT * fast_tanh(acc);
        logits[t] = (maskw[(size_t)b * SDIM + t] > 0.f) ? NEG_INFF : lg;
    }
    __syncthreads();

    // ---- max reduce
    red[t] = logits[t];
    __syncthreads();
    for (int off = SDIM / 2; off > 0; off >>= 1) {
        if (t < off) red[t] = fmaxf(red[t], red[t + off]);
        __syncthreads();
    }
    float mx = red[0];
    __syncthreads();

    // ---- log-sum-exp
    red[t] = __expf(logits[t] - mx);
    __syncthreads();
    for (int off = SDIM / 2; off > 0; off >>= 1) {
        if (t < off) red[t] += red[t + off];
        __syncthreads();
    }
    float lse = mx + __logf(red[0]);
    __syncthreads();

    // ---- Gumbel perturbation (deterministic hash RNG) + argmax
    unsigned h32 = (unsigned)step * 0x9E3779B9u ^ (unsigned)b * 0x85EBCA6Bu
                 ^ (unsigned)t * 0xC2B2AE35u;
    h32 ^= h32 >> 16; h32 *= 0x7FEB352Du;
    h32 ^= h32 >> 15; h32 *= 0x846CA68Bu;
    h32 ^= h32 >> 16;
    float uu = ((h32 >> 8) + 0.5f) * (1.0f / 16777216.0f);
    float g  = -__logf(-__logf(uu));
    red[t] = logits[t] + g; redi[t] = t;
    __syncthreads();
    for (int off = SDIM / 2; off > 0; off >>= 1) {
        if (t < off) {
            if (red[t + off] > red[t] ||
                (red[t + off] == red[t] && redi[t + off] < redi[t])) {
                red[t] = red[t + off]; redi[t] = redi[t + off];
            }
        }
        __syncthreads();
    }
    if (t == 0) {
        sel = redi[0];
        out_idx[(size_t)b * (SDIM - 1) + step] = (float)redi[0];
        out_lp [(size_t)b * (SDIM - 1) + step] = logits[redi[0]] - lse;
        maskw[(size_t)b * SDIM + redi[0]] = 1.0f;
    }
    __syncthreads();

    // ---- gather picked embedding, update init_h (step 0) and query (bf16)
    int idx = sel;
    if (t < EDIM) {
        float h = cv[((size_t)b * SDIM + idx) * EDIM + t];
        hsel[t] = h;
        float ih = (step == 0) ? h : inith[(size_t)b * EDIM + t];
        if (step == 0) inith[(size_t)b * EDIM + t] = h;
        ihrow[t] = ih;
    }
    __syncthreads();
    if (t < EDIM) {
        float acc = hbar[(size_t)b * EDIM + t] + bv[t];
        for (int j = 0; j < EDIM; ++j) {
            acc = fmaf(ihrow[j], Wv[j * EDIM + t], acc);
            acc = fmaf(hsel[j],  Wv[(EDIM + j) * EDIM + t], acc);
        }
        qbf[(size_t)b * EDIM + t] = f2bf(acc);
    }
}

// ---------------------------------------------------------------------------
extern "C" void kernel_launch(void* const* d_in, const int* in_sizes, int n_in,
                              void* d_out, int out_size, void* d_ws, size_t ws_size,
                              hipStream_t stream) {
    const float* cv   = (const float*)d_in[0];   // [B,S,E]
    // d_in[1] original_node: unused by reference outputs
    const float* mask = (const float*)d_in[2];   // [B,S]
    const float* liw  = (const float*)d_in[3];   // [2E]
    const float* Wc   = (const float*)d_in[4];   // [E,E]
    const float* bc   = (const float*)d_in[5];   // [E]
    const float* Wv   = (const float*)d_in[6];   // [2E,E]
    const float* bv   = (const float*)d_in[7];   // [E]
    const float* Wq   = (const float*)d_in[8];   // [E,H]
    const float* Wref = (const float*)d_in[9];   // [E,H]
    const float* v    = (const float*)d_in[10];  // [H]

    char* ws = (char*)d_ws;
    size_t off = 0;
    auto take = [&](size_t bytes) {
        size_t o = off;
        off += (bytes + 255) & ~(size_t)255;
        return o;
    };
    float*  ref   = (float*)(ws + take((size_t)BDIM * SDIM * HDIM * 4));  // 64 MB
    __bf16* cvbf  = (__bf16*)(ws + take((size_t)BDIM * SDIM * EDIM * 2)); // 32 MB
    __bf16* wreff = (__bf16*)(ws + take((size_t)16384 * 2));              // 32 KB
    __bf16* wqf   = (__bf16*)(ws + take((size_t)16384 * 2));              // 32 KB
    __bf16* qbf   = (__bf16*)(ws + take((size_t)BDIM * EDIM * 2));
    float*  hbar  = (float*)(ws + take((size_t)BDIM * EDIM * 4));
    float*  qWb   = (float*)(ws + take((size_t)BDIM * HDIM * 4));
    float*  inith = (float*)(ws + take((size_t)BDIM * EDIM * 4));
    float*  maskw = (float*)(ws + take((size_t)BDIM * SDIM * 4));

    float* out_idx = (float*)d_out;
    float* out_lp  = out_idx + (size_t)BDIM * (SDIM - 1);

    // ---- one-time precompute
    k_mean_hbar<<<BDIM, EDIM, 0, stream>>>(cv, Wc, bc, hbar);
    k_init<<<BDIM, SDIM, 0, stream>>>(mask, liw, Wv, bv, hbar, maskw, qbf, inith);
    {   // bf16 conversions + weight fragment packing
        size_t n4 = (size_t)BDIM * SDIM * EDIM / 4;
        k_cvt_bf16<<<(unsigned)(n4 / 256), 256, 0, stream>>>((const float4*)cv,
                                                             (v4bf*)cvbf);
        k_packW<<<64, 256, 0, stream>>>(Wref, wreff);
        k_packW<<<64, 256, 0, stream>>>(Wq, wqf);
    }
    // ref = cv @ Wref : [B*S,128] x [128,128] -> L2-resident (64 MB)
    k_gemm_bf16<<<(BDIM * SDIM) / 128, 256, 0, stream>>>(cvbf, wreff, ref,
                                                         BDIM * SDIM);

    // ---- 255 dependent decode steps (graph-capturable kernel chain)
    for (int t = 0; t < SDIM - 1; ++t) {
        k_gemm_bf16<<<BDIM / 128, 256, 0, stream>>>(qbf, wqf, qWb, BDIM);
        k_step<<<BDIM, SDIM, 0, stream>>>(cv, ref, qWb, v, hbar, Wv, bv,
                                          maskw, qbf, inith,
                                          out_idx, out_lp, t);
    }
}
